// RNN_ATT_noise_22651657519485
// MI455X (gfx1250) — matrix operand order repaired and checked by reading, not compiled, not run on hardware
//
#include <hip/hip_runtime.h>
#include <math.h>

// Problem constants (from reference): B=256, C=8, T=512, H=512, OUT=2
#define BB 256
#define TT 512
#define HH 512
#define HPAD 520   // LDS row stride in elements: 520*2B = 1040 B = 65*16 B
                   // -> 16B-aligned rows, lanes hit distinct LDS banks

typedef __attribute__((ext_vector_type(16))) __bf16 v16bf;
typedef __attribute__((ext_vector_type(8)))  __bf16 v8bf;
typedef __attribute__((ext_vector_type(8)))  float  v8f;

// ---------------------------------------------------------------------------
// Init: W_rec -> bf16, fused bias, zero t=0 output slices, tm_modified copy,
// prs_1 (carry key0 == 1).  (inp_1 is computed inside the persistent kernel.)
// ---------------------------------------------------------------------------
__global__ __launch_bounds__(256) void rnn_init_kernel(
    const float* __restrict__ inputs,  // (B,8,T)
    const float* __restrict__ W_rec,   // (H,H)
    const float* __restrict__ b_in,    // (H)
    const float* __restrict__ b_rec,   // (H)
    float* __restrict__ keys_out,      // (B,2,T)
    float* __restrict__ prs_out,       // (B,4,T)
    float* __restrict__ hs_out,        // (B,H,T)
    float* __restrict__ tm_out,        // (B,2,T)
    __bf16* __restrict__ wrec_bf,      // (H,H) bf16
    float* __restrict__ bias)          // (H) = b_in + b_rec
{
  const int gsz = gridDim.x * blockDim.x;
  const int tid = blockIdx.x * blockDim.x + threadIdx.x;

  for (int i = tid; i < HH * HH; i += gsz)
    wrec_bf[i] = (__bf16)W_rec[i];

  for (int i = tid; i < BB * HH; i += gsz) {
    int b = i / HH, j = i % HH;
    hs_out[(size_t)b * HH * TT + (size_t)j * TT + 0] = 0.0f;
  }

  for (int i = tid; i < BB * 2 * TT; i += gsz) {
    int b = i / (2 * TT);
    int r = i % (2 * TT);
    int c = r / TT;
    int t = r % TT;
    __builtin_nontemporal_store(inputs[(size_t)b * 8 * TT + (size_t)(6 + c) * TT + t],
                                &tm_out[i]);
  }

  for (int i = tid; i < HH; i += gsz)
    bias[i] = b_in[i] + b_rec[i];

  for (int b = tid; b < BB; b += gsz) {
    keys_out[(size_t)b * 2 * TT + 0]  = 0.0f;
    keys_out[(size_t)b * 2 * TT + TT] = 0.0f;
    for (int p = 0; p < 4; ++p)
      prs_out[(size_t)b * 4 * TT + (size_t)p * TT + 0] = 0.0f;

    // prs at t=1 (key0 == 1 -> arms_sum = arms_sec[:, :2])
    const float* xb = inputs + (size_t)b * 8 * TT;
    const int tn = 1;
    float a0  = xb[0 * TT + tn], a1 = xb[1 * TT + tn];
    float s0  = xb[2 * TT + tn], s1 = xb[3 * TT + tn];
    float tm0 = xb[6 * TT + tn], tm1 = xb[7 * TT + tn];
    float r;
    r = (tm0 - a0) / (0.15f * a0); prs_out[(size_t)b*4*TT + 0*TT + tn] = (tm0 == 0.0f) ? 0.0f : r * r;
    r = (tm0 - a1) / (0.15f * a1); prs_out[(size_t)b*4*TT + 1*TT + tn] = (tm0 == 0.0f) ? 0.0f : r * r;
    r = (tm1 - s0) / (0.15f * s0); prs_out[(size_t)b*4*TT + 2*TT + tn] = (tm1 == 0.0f) ? 0.0f : r * r;
    r = (tm1 - s1) / (0.15f * s1); prs_out[(size_t)b*4*TT + 3*TT + tn] = (tm1 == 0.0f) ? 0.0f : r * r;
  }
}

// ---------------------------------------------------------------------------
// Persistent per-batch-slice kernel. The recurrence is block-diagonal over
// batch rows: one 1024-thread workgroup (32 waves) owns 16 batch rows and all
// 512 hidden columns, iterating all 511 steps with workgroup barriers only.
// h state lives in LDS (ping/pong bf16 + f32 for the head).
//   phase 1: wave w computes a 16x16 tile with 16x v_wmma_f32_16x16x32_bf16;
//            A from LDS (bank-conflict-free via HPAD), B via global_load each
//            iteration (laundered zero offset keeps global provenance while
//            preventing cross-t register caching -> no spills, no flat ops).
//   phase 2: waves 0..15: key = sigmoid(h @ W_key^T + b_key) for one row,
//            then inp_{t+1} (to LDS) and prs_{t+1}.
// Grid: 16 blocks x 1024 threads; no cross-block communication.
// ---------------------------------------------------------------------------
__global__ __launch_bounds__(1024) void rnn_persistent(
    const float* __restrict__ inputs,  // (B,8,T)
    const float* __restrict__ W_in,    // (H,6)
    const float* __restrict__ W_key,   // (2,H)
    const float* __restrict__ b_key,   // (2)
    const float* __restrict__ bias,    // (H)
    const __bf16* __restrict__ wrec,   // (H,H) bf16 row-major
    float* __restrict__ keys_out,      // (B,2,T)
    float* __restrict__ prs_out,       // (B,4,T)
    float* __restrict__ hs_out)        // (B,H,T)
{
  __shared__ __bf16 sh_h[2][16 * HPAD];   // ping/pong h (bf16), 2 x 16.25 KB
  __shared__ float  sh_hf[16 * HPAD];     // h (f32) for the head, 32.5 KB
  __shared__ float  sh_inp[16 * 6];       // per-row 6-wide GEMM input

  const int tidx  = threadIdx.x;
  const int lane  = tidx & 31;
  const int wv    = tidx >> 5;            // 0..31 = hidden-column tile
  const int kh    = lane >> 4;
  const int l15   = lane & 15;
  const int mbase = blockIdx.x * 16;      // batch rows owned by this block
  const int ncol  = wv * 16 + l15;        // output column (C/D layout)

  // t-invariant input-GEMM weights / bias for this lane's column
  const float w0 = W_in[ncol*6+0], w1 = W_in[ncol*6+1], w2 = W_in[ncol*6+2];
  const float w3 = W_in[ncol*6+3], w4 = W_in[ncol*6+4], w5 = W_in[ncol*6+5];
  const float bn = bias[ncol];
  const float bk0 = b_key[0], bk1 = b_key[1];
  const float* xh = inputs + (size_t)(mbase + wv) * 8 * TT;  // head wave's row

  // h_0 = 0 in ping buffer
  for (int i = tidx; i < 16 * HPAD; i += 1024)
    sh_h[0][i] = (__bf16)0.0f;
  // inp_1 (carry key0 == 1 -> arms_sum = arms_sec[:, :2])
  if (tidx < 16) {
    const float* xb = inputs + (size_t)(mbase + tidx) * 8 * TT;
    float* ip = &sh_inp[tidx * 6];
    ip[0] = xb[0 * TT + 1];
    ip[1] = xb[1 * TT + 1];
    ip[2] = xb[2 * TT + 1];
    ip[3] = xb[3 * TT + 1];
    ip[4] = xb[6 * TT + 1];
    ip[5] = xb[7 * TT + 1];
  }
  __syncthreads();

  const __bf16* bp = wrec + (size_t)ncol * HH + kh * 16;  // B base (invariant)

  for (int t = 1; t < TT; ++t) {
    const __bf16* hsrc = sh_h[(t & 1) ^ 1];   // t=1 reads buffer 0 (h_0)
    __bf16*       hdst = sh_h[t & 1];

    // ---- phase 1: recurrent GEMM tile ----
    v8f acc;
#pragma unroll
    for (int v = 0; v < 8; ++v) {
      const float* ip = &sh_inp[(kh * 8 + v) * 6];
      acc[v] = bn + ip[0]*w0 + ip[1]*w1 + ip[2]*w2 + ip[3]*w3 + ip[4]*w4 + ip[5]*w5;
    }

    const __bf16* ap = hsrc + l15 * HPAD + kh * 16;  // LDS A base

    // Launder a zero offset: base keeps global provenance (-> global_load,
    // not flat_load), but the asm-defined value each iteration stops the
    // compiler from register-caching the loop-invariant B fragments.
    unsigned boff = 0;
    asm volatile("" : "+v"(boff));
    const __bf16* bpt = bp + boff;

    // 1-deep software pipeline over the 16 K-steps
    v16bf acur, bcur;
    {
      v8bf a0 = *(const v8bf*)(ap);
      v8bf a1 = *(const v8bf*)(ap + 8);
      acur = __builtin_shufflevector(a0, a1,
               0,1,2,3,4,5,6,7,8,9,10,11,12,13,14,15);
      bcur = *(const v16bf*)(bpt);
    }
#pragma unroll
    for (int i = 0; i < 15; ++i) {
      v8bf a0n = *(const v8bf*)(ap + (i + 1) * 32);
      v8bf a1n = *(const v8bf*)(ap + (i + 1) * 32 + 8);
      v16bf anext = __builtin_shufflevector(a0n, a1n,
                      0,1,2,3,4,5,6,7,8,9,10,11,12,13,14,15);
      v16bf bnext = *(const v16bf*)(bpt + (i + 1) * 32);
      acc = __builtin_amdgcn_wmma_f32_16x16x32_bf16(
          false, acur, false, bcur, (short)0, acc, false, false);
      acur = anext;
      bcur = bnext;
    }
    acc = __builtin_amdgcn_wmma_f32_16x16x32_bf16(
        false, acur, false, bcur, (short)0, acc, false, false);

#pragma unroll
    for (int v = 0; v < 8; ++v) {
      int mr = kh * 8 + v;                        // local row 0..15
      float hv = tanhf(acc[v]);
      sh_hf[mr * HPAD + ncol] = hv;
      hdst[mr * HPAD + ncol]  = (__bf16)hv;
      __builtin_nontemporal_store(hv,
          &hs_out[(size_t)(mbase + mr) * HH * TT + (size_t)ncol * TT + t]);
    }

    __syncthreads();   // h_t complete in LDS

    // ---- phase 2: head (waves 0..15, one batch row each) ----
    if (wv < 16) {
      // Same laundering for W_key: WGP$-hit global loads each step instead of
      // pinning 32 VGPRs in every wave.
      unsigned wkoff = 0;
      asm volatile("" : "+v"(wkoff));
      const float* wkp = W_key + wkoff;

      const float* hrow = &sh_hf[wv * HPAD];
      float d0 = 0.0f, d1 = 0.0f;
#pragma unroll
      for (int i = 0; i < 16; ++i) {
        float hv = hrow[lane + i * 32];
        d0 += hv * wkp[lane + i * 32];
        d1 += hv * wkp[HH + lane + i * 32];
      }
#pragma unroll
      for (int off = 16; off >= 1; off >>= 1) {
        d0 += __shfl_xor(d0, off, 32);
        d1 += __shfl_xor(d1, off, 32);
      }

      if (lane == 0) {
        const int b = mbase + wv;
        float k0 = 1.0f / (1.0f + expf(-(d0 + bk0)));
        float k1 = 1.0f / (1.0f + expf(-(d1 + bk1)));
        keys_out[(size_t)b * 2 * TT + t]      = k0;
        keys_out[(size_t)b * 2 * TT + TT + t] = k1;

        if (t + 1 < TT) {
          const int tn = t + 1;
          float a0  = xh[0 * TT + tn], a1 = xh[1 * TT + tn];
          float s0  = xh[2 * TT + tn], s1 = xh[3 * TT + tn];
          float s2  = xh[4 * TT + tn], s3 = xh[5 * TT + tn];
          float tm0 = xh[6 * TT + tn], tm1 = xh[7 * TT + tn];
          float as0 = s0 * k0 + s2 * (1.0f - k0);
          float as1 = s1 * k0 + s3 * (1.0f - k0);
          float* ip = &sh_inp[wv * 6];
          ip[0] = a0; ip[1] = a1; ip[2] = as0; ip[3] = as1; ip[4] = tm0; ip[5] = tm1;

          float r;
          r = (tm0 - a0)  / (0.15f * a0);  prs_out[(size_t)b*4*TT + 0*TT + tn] = (tm0 == 0.0f) ? 0.0f : r * r;
          r = (tm0 - a1)  / (0.15f * a1);  prs_out[(size_t)b*4*TT + 1*TT + tn] = (tm0 == 0.0f) ? 0.0f : r * r;
          r = (tm1 - as0) / (0.15f * as0); prs_out[(size_t)b*4*TT + 2*TT + tn] = (tm1 == 0.0f) ? 0.0f : r * r;
          r = (tm1 - as1) / (0.15f * as1); prs_out[(size_t)b*4*TT + 3*TT + tn] = (tm1 == 0.0f) ? 0.0f : r * r;
        }
      }
    }

    __syncthreads();   // inp_{t+1} visible before next GEMM
  }
}

// ---------------------------------------------------------------------------
// Host orchestration: exactly two launches, both on `stream`.
// ---------------------------------------------------------------------------
extern "C" void kernel_launch(void* const* d_in, const int* in_sizes, int n_in,
                              void* d_out, int out_size, void* d_ws, size_t ws_size,
                              hipStream_t stream) {
  const float* inputs = (const float*)d_in[0];  // (B,8,T)
  const float* W_in   = (const float*)d_in[1];  // (H,6)
  const float* b_in   = (const float*)d_in[2];  // (H)
  const float* W_rec  = (const float*)d_in[3];  // (H,H)
  const float* b_rec  = (const float*)d_in[4];  // (H)
  const float* W_key  = (const float*)d_in[5];  // (2,H)
  const float* b_key  = (const float*)d_in[6];  // (2)

  float* out = (float*)d_out;
  float* keys_out = out;                                               // B*2*T
  float* prs_out  = out + (size_t)BB * 2 * TT;                         // B*4*T
  float* hs_out   = out + (size_t)BB * 2 * TT + (size_t)BB * 4 * TT;   // B*H*T
  float* tm_out   = hs_out + (size_t)BB * HH * TT;                     // B*2*T

  // Workspace: only converted weights + fused bias.
  char* ws = (char*)d_ws;
  __bf16* wrec_bf = (__bf16*)(ws + 0);        // H*H*2 = 524288
  float*  bias    = (float*) (ws + 524288);   // H*4   = 2048

  rnn_init_kernel<<<512, 256, 0, stream>>>(
      inputs, W_rec, b_in, b_rec,
      keys_out, prs_out, hs_out, tm_out,
      wrec_bf, bias);

  rnn_persistent<<<BB / 16, 1024, 0, stream>>>(
      inputs, W_in, W_key, b_key, bias, wrec_bf,
      keys_out, prs_out, hs_out);
}
